// MultiHeadAttentionBlock_49967649521921
// MI455X (gfx1250) — matrix operand order repaired and checked by reading, not compiled, not run on hardware
//
#include <hip/hip_runtime.h>

// ---------------------------------------------------------------------------
// MultiHeadAttentionBlock for MI455X (gfx1250, wave32, WMMA 16x16x32 f16->f32)
//
// Pipeline (f16 compute, f32 accumulate, f32 output):
//   0. convert Wq/Wk/Wv/Wo f32->f16;  Xt[b][n][c] = (f16) x[b][c][n]  (LDS transpose)
//   1. Qh[b][o][n] = Wq . Xt^T   (NT GEMM, bias per-row)      [512 x 1024, K=512]
//      Kh[b][o][n] = Wk . Xt^T
//      Vt[b][n][o] = Xt . Wv^T   (NT GEMM, bias per-col)      [1024 x 512, K=512]
//   2. per (b,h): scores = Qh.Kh^T / 8 ; softmax ; attn.Vt written straight into
//      the torch transpose+reshape layout ORt[p][c] (p=(n&15)*64+d, c=h*64+(n>>4))
//   3. y[b][o][p] = Wo . ORt^T + bo   -> d_out (f32)
//
// Every GEMM is "NT" (both operands row-major over K) so all WMMA fragment
// loads are contiguous b128. GEMM waves use 32x64 tiles: 8 WMMAs per 12 b128
// loads (~21 FLOP/B vs ~13 at 16x64), since the projections dominate FLOPs.
// ---------------------------------------------------------------------------

typedef __attribute__((ext_vector_type(16))) _Float16 v16h;
typedef __attribute__((ext_vector_type(8)))  float    v8f;
typedef __attribute__((ext_vector_type(4)))  float    f32x4;

static constexpr int BATCH = 16;
static constexpr int CDIM  = 512;   // channels
static constexpr int NPIX  = 1024;  // 32*32
static constexpr int NH    = 8;
static constexpr int HD    = 64;

// ---- WMMA fragment loaders (contiguous k order, same for A and B) ----
// A 16x32 (MxK): lane<16 -> M=lane, K=k0..k0+15;
//                lane>=16 -> same M rows, K=k0+16..k0+31.
__device__ __forceinline__ v16h load_frag_a(const _Float16* base, int m0, int k0, int ld) {
  int lane = threadIdx.x & 31;
  const _Float16* p = base + (size_t)(m0 + (lane & 15)) * ld + k0 + ((lane >> 4) << 4);
  union { v16h v; f32x4 q[2]; } u;
  u.q[0] = *(const f32x4*)p;
  u.q[1] = *(const f32x4*)(p + 8);
  return u.v;
}
// B 32x16 (KxN): lane<16 -> N=lane, K=k0..k0+15 ; lane>=16 -> same N, K=k0+16..31.
// "base" is B^T row-major: row = output column, contiguous over K.
__device__ __forceinline__ v16h load_frag_b(const _Float16* base, int n0, int k0, int ld) {
  int lane = threadIdx.x & 31;
  const _Float16* p = base + (size_t)(n0 + (lane & 15)) * ld + k0 + ((lane >> 4) << 4);
  union { v16h v; f32x4 q[2]; } u;
  u.q[0] = *(const f32x4*)p;
  u.q[1] = *(const f32x4*)(p + 8);
  return u.v;
}

__device__ __forceinline__ v8f wmma_f16(v16h a, v16h b, v8f c) {
  return __builtin_amdgcn_wmma_f32_16x16x32_f16(false, a, false, b, (short)0, c, false, false);
}

// D 16x16 f32: VGPR r -> (M = r + 8*(lane>=16), N = lane&15)
template <int BIAS_MODE /*0=none,1=per-row,2=per-col*/, bool F16OUT>
__device__ __forceinline__ void store_tile(void* dst, int ld, int m0, int n0,
                                           v8f acc, const float* bias) {
  int lane = threadIdx.x & 31;
  int col  = n0 + (lane & 15);
  int hi   = (lane >> 4) << 3;  // 0 or 8
  float bc = (BIAS_MODE == 2) ? bias[col] : 0.0f;
#pragma unroll
  for (int r = 0; r < 8; ++r) {
    int row  = m0 + r + hi;
    float v  = acc[r];
    if (BIAS_MODE == 1) v += bias[row];
    if (BIAS_MODE == 2) v += bc;
    if (F16OUT) ((_Float16*)dst)[(size_t)row * ld + col] = (_Float16)v;
    else        ((float*)dst)[(size_t)row * ld + col]    = v;
  }
}

// ---------------- universal NT GEMM: D[M][N] = A[M][K] * B[N][K]^T ----------
// 128 threads = 4 waves; workgroup tile 128x64; wave w owns rows [32w,32w+32):
// two A fragments amortize each B fragment -> 8 WMMA per 12 b128 loads.
template <int BIAS_MODE, bool F16OUT>
__global__ __launch_bounds__(128) void gemm_nt(
    const _Float16* __restrict__ A, size_t strideA, int lda,
    const _Float16* __restrict__ B, size_t strideB, int ldb,
    void* __restrict__ D, size_t strideD, int ldd,
    const float* __restrict__ bias, int K) {
  int b = blockIdx.z;
  const _Float16* Ab = A + (size_t)b * strideA;
  const _Float16* Bb = B + (size_t)b * strideB;
  void* Db = F16OUT ? (void*)((_Float16*)D + (size_t)b * strideD)
                    : (void*)((float*)D + (size_t)b * strideD);
  int wave = threadIdx.x >> 5;
  int m0 = blockIdx.x * 128 + wave * 32;
  int n0 = blockIdx.y * 64;

  v8f acc[2][4] = {};
  for (int k0 = 0; k0 < K; k0 += 32) {
    v16h a0 = load_frag_a(Ab, m0,      k0, lda);
    v16h a1 = load_frag_a(Ab, m0 + 16, k0, lda);
#pragma unroll
    for (int j = 0; j < 4; ++j) {
      v16h bf = load_frag_b(Bb, n0 + 16 * j, k0, ldb);
      acc[0][j] = wmma_f16(a0, bf, acc[0][j]);
      acc[1][j] = wmma_f16(a1, bf, acc[1][j]);
    }
  }
#pragma unroll
  for (int j = 0; j < 4; ++j) {
    store_tile<BIAS_MODE, F16OUT>(Db, ldd, m0,      n0 + 16 * j, acc[0][j], bias);
    store_tile<BIAS_MODE, F16OUT>(Db, ldd, m0 + 16, n0 + 16 * j, acc[1][j], bias);
  }
}

// ---------------- fused attention: scores -> softmax -> attn.V --------------
// One workgroup per (head, batch). 128 threads = 4 waves, 16-row strips.
__global__ __launch_bounds__(128) void attn_kernel(
    const _Float16* __restrict__ Qh, const _Float16* __restrict__ Kh,
    const _Float16* __restrict__ Vt, _Float16* __restrict__ ORt) {
  int h = blockIdx.x, b = blockIdx.y;
  int wave = threadIdx.x >> 5;
  int lane = threadIdx.x & 31;
  int hi8  = (lane >> 4) << 3;

  const _Float16* Qb = Qh + ((size_t)b * CDIM + h * HD) * NPIX;  // [d][n], ld=1024
  const _Float16* Kb = Kh + ((size_t)b * CDIM + h * HD) * NPIX;  // [e][n], ld=1024
  const _Float16* Vb = Vt + (size_t)b * NPIX * CDIM + h * HD;    // Vt[n][h*64+e]
  _Float16*       Ob = ORt + (size_t)b * NPIX * CDIM + h * HD;   // ORt[p][h*64+q]

  __shared__ float    sS[64][65];  // scores f32 (stride 65 -> conflict-free)
  __shared__ _Float16 sA[64][72];  // attn  f16 (stride 72 -> conflict-free b128)

  // scores[d][e] = (1/8) * sum_n Q[d][n] K[e][n]   (NT GEMM, K=1024)
  v8f sc[4] = {};
  for (int k0 = 0; k0 < NPIX; k0 += 32) {
    v16h a = load_frag_a(Qb, 16 * wave, k0, NPIX);
#pragma unroll
    for (int j = 0; j < 4; ++j)
      sc[j] = wmma_f16(a, load_frag_b(Kb, 16 * j, k0, NPIX), sc[j]);
  }
#pragma unroll
  for (int j = 0; j < 4; ++j)
#pragma unroll
    for (int r = 0; r < 8; ++r)
      sS[16 * wave + r + hi8][16 * j + (lane & 15)] = sc[j][r] * 0.125f;
  __syncthreads();

  // row-wise softmax over e (64 rows, one thread each)
  if (threadIdx.x < 64) {
    int d = threadIdx.x;
    float mx = -3.0e38f;
    for (int e = 0; e < 64; ++e) mx = fmaxf(mx, sS[d][e]);
    float sum = 0.0f;
    for (int e = 0; e < 64; ++e) { float t = __expf(sS[d][e] - mx); sS[d][e] = t; sum += t; }
    float inv = 1.0f / sum;
    for (int e = 0; e < 64; ++e) sA[d][e] = (_Float16)(sS[d][e] * inv);
  }
  __syncthreads();

  // attn.V in 16 per-s blocks: out(s)[d][q] = sum_e attn[d][e] * Vt[16q+s][h*64+e]
  // lands directly in torch's transpose+reshape layout: ORt[s*64+d][h*64+q].
  for (int s = 0; s < 16; ++s) {
    v8f o[4] = {};
#pragma unroll
    for (int e0 = 0; e0 < HD; e0 += 32) {
      v16h a = load_frag_a(&sA[0][0], 16 * wave, e0, 72);
#pragma unroll
      for (int j = 0; j < 4; ++j) {
        int q = 16 * j + (lane & 15);
        const _Float16* p = Vb + (size_t)(16 * q + s) * CDIM + e0 + ((lane >> 4) << 4);
        union { v16h v; f32x4 x[2]; } u;
        u.x[0] = *(const f32x4*)p;
        u.x[1] = *(const f32x4*)(p + 8);
        o[j] = wmma_f16(a, u.v, o[j]);
      }
    }
#pragma unroll
    for (int j = 0; j < 4; ++j)
#pragma unroll
      for (int r = 0; r < 8; ++r) {
        int d = 16 * wave + r + hi8;
        int q = 16 * j + (lane & 15);
        Ob[(size_t)(s * 64 + d) * CDIM + q] = (_Float16)o[j][r];
      }
  }
}

// ---------------- data-prep kernels ----------------------------------------
__global__ __launch_bounds__(256) void convert_f32_to_f16(
    const float* __restrict__ src, _Float16* __restrict__ dst, int n) {
  int i = blockIdx.x * 256 + threadIdx.x;
  if (i < n) dst[i] = (_Float16)src[i];
}

// Xt[b][n][c] = (f16) x[b][c][n] via 32x32 LDS tile transpose
__global__ __launch_bounds__(256) void transpose_convert_x(
    const float* __restrict__ x, _Float16* __restrict__ Xt) {
  __shared__ float tile[32][33];
  int n0 = blockIdx.x * 32, c0 = blockIdx.y * 32, b = blockIdx.z;
  const float* xb = x + (size_t)b * CDIM * NPIX;
  _Float16* xtb = Xt + (size_t)b * NPIX * CDIM;
  int tx = threadIdx.x & 31, ty = threadIdx.x >> 5;  // 32 x 8
  for (int i = ty; i < 32; i += 8)
    tile[i][tx] = xb[(size_t)(c0 + i) * NPIX + n0 + tx];
  __syncthreads();
  for (int i = ty; i < 32; i += 8)
    xtb[(size_t)(n0 + i) * CDIM + c0 + tx] = (_Float16)tile[tx][i];
}

// ---------------------------------------------------------------------------
extern "C" void kernel_launch(void* const* d_in, const int* in_sizes, int n_in,
                              void* d_out, int out_size, void* d_ws, size_t ws_size,
                              hipStream_t stream) {
  const float* x  = (const float*)d_in[0];
  const float* Wq = (const float*)d_in[1];
  const float* bq = (const float*)d_in[2];
  const float* Wk = (const float*)d_in[3];
  const float* bk = (const float*)d_in[4];
  const float* Wv = (const float*)d_in[5];
  const float* bv = (const float*)d_in[6];
  const float* Wo = (const float*)d_in[7];
  const float* bo = (const float*)d_in[8];
  float* y = (float*)d_out;

  const size_t WSZ = (size_t)CDIM * CDIM;            // 262144
  const size_t TSZ = (size_t)BATCH * CDIM * NPIX;    // 8388608
  _Float16* Wqh = (_Float16*)d_ws;
  _Float16* Wkh = Wqh + WSZ;
  _Float16* Wvh = Wkh + WSZ;
  _Float16* Woh = Wvh + WSZ;
  _Float16* Qh  = Woh + WSZ;
  _Float16* Kh  = Qh + TSZ;
  _Float16* Vt  = Kh + TSZ;
  _Float16* Xt  = Vt + TSZ;       // reused as ORt after the projections
  _Float16* ORt = Xt;             // total scratch ~66 MB

  // 0) weight + activation conversion
  dim3 cb(256);
  convert_f32_to_f16<<<dim3((WSZ + 255) / 256), cb, 0, stream>>>(Wq, Wqh, (int)WSZ);
  convert_f32_to_f16<<<dim3((WSZ + 255) / 256), cb, 0, stream>>>(Wk, Wkh, (int)WSZ);
  convert_f32_to_f16<<<dim3((WSZ + 255) / 256), cb, 0, stream>>>(Wv, Wvh, (int)WSZ);
  convert_f32_to_f16<<<dim3((WSZ + 255) / 256), cb, 0, stream>>>(Wo, Woh, (int)WSZ);
  transpose_convert_x<<<dim3(NPIX / 32, CDIM / 32, BATCH), cb, 0, stream>>>(x, Xt);

  const size_t sAct = (size_t)NPIX * CDIM;  // per-batch activation stride

  // 1) Q[b][o][n] = Wq . Xt^T + bq   (M=512, N=1024, K=512)
  gemm_nt<1, true><<<dim3(CDIM / 128, NPIX / 64, BATCH), dim3(128), 0, stream>>>(
      Wqh, 0, CDIM, Xt, sAct, CDIM, Qh, sAct, NPIX, bq, CDIM);
  gemm_nt<1, true><<<dim3(CDIM / 128, NPIX / 64, BATCH), dim3(128), 0, stream>>>(
      Wkh, 0, CDIM, Xt, sAct, CDIM, Kh, sAct, NPIX, bk, CDIM);
  //    Vt[b][n][o] = Xt . Wv^T + bv  (M=1024, N=512, K=512; bias on N)
  gemm_nt<2, true><<<dim3(NPIX / 128, CDIM / 64, BATCH), dim3(128), 0, stream>>>(
      Xt, sAct, CDIM, Wvh, 0, CDIM, Vt, sAct, CDIM, bv, CDIM);

  // 2) fused attention per (head, batch)
  attn_kernel<<<dim3(NH, BATCH), dim3(128), 0, stream>>>(Qh, Kh, Vt, ORt);

  // 3) y[b][o][p] = Wo . ORt^T + bo  (f32 out, straight into d_out)
  gemm_nt<1, false><<<dim3(CDIM / 128, NPIX / 64, BATCH), dim3(128), 0, stream>>>(
      Woh, 0, CDIM, ORt, sAct, CDIM, y, sAct, NPIX, bo, CDIM);

  (void)in_sizes; (void)n_in; (void)out_size; (void)ws_size;
}
